// MSA_54211077210188
// MI455X (gfx1250) — matrix operand order repaired and checked by reading, not compiled, not run on hardware
//
#include <hip/hip_runtime.h>
#include <hip/hip_bf16.h>

// ---------------------------------------------------------------------------
// MSA windowed attention, CDNA5 (gfx1250) bf16 WMMA implementation.
// Bn=512, W=128, D=768, H=12, hd=64, f*w=128, scale=0.125, prune=0.5
// Async global->LDS staging (ASYNCcnt) for all workgroup-shared operands.
// ---------------------------------------------------------------------------

typedef __bf16 bf16_t;
typedef __attribute__((ext_vector_type(16))) __bf16 v16bf;
typedef __attribute__((ext_vector_type(8)))  __bf16 v8bf;
typedef __attribute__((ext_vector_type(8)))  float  v8f;

#define BN     512
#define WTOK   128
#define DMODEL 768
#define NHEAD  12
#define HD     64
#define QKV_N  2304
#define FW     128

__device__ inline bf16_t f2bf(float f) {
    unsigned int u = __float_as_uint(f);
    unsigned int r = u + 0x7FFFu + ((u >> 16) & 1u);   // round-to-nearest-even
    unsigned short h = (unsigned short)(r >> 16);
    return __builtin_bit_cast(__bf16, h);
}

// Assemble a 16x32 bf16 A-fragment: elements 0..7 from lo ptr, 8..15 from hi ptr
// (per-lane contiguous 8-element K segments at k 16h and 16h+8: contiguous k = 16h + s, as B uses).
__device__ inline v16bf ld_a_pair(const bf16_t* lo, const bf16_t* hi) {
    v8bf a = *(const v8bf*)lo;
    v8bf b = *(const v8bf*)hi;
    return __builtin_shufflevector(a, b, 0,1,2,3,4,5,6,7,8,9,10,11,12,13,14,15);
}

// Async copy of one 16-byte chunk from global memory into LDS (per lane).
// GLOBAL_LOAD_ASYNC_TO_LDS_B128: VDST = LDS byte offset, VADDR = 64-bit addr.
// Generic LDS pointers truncate to the 32-bit LDS offset (aperture rule).
__device__ inline void async_g2l_b128(const bf16_t* g, bf16_t* l) {
    unsigned lofs = (unsigned)(unsigned long long)l;
    asm volatile("global_load_async_to_lds_b128 %0, %1, off"
                 :: "v"(lofs), "v"(g) : "memory");
}

__device__ inline void wait_async0() {
    asm volatile("s_wait_asynccnt 0x0" ::: "memory");
}
__device__ inline void wait_async2() {
    asm volatile("s_wait_asynccnt 0x2" ::: "memory");
}

// ---------------------------------------------------------------------------
// f32 -> bf16 elementwise convert
// ---------------------------------------------------------------------------
__global__ void cvt_f32_bf16(const float* __restrict__ in, bf16_t* __restrict__ out, int n) {
    int i = blockIdx.x * blockDim.x + threadIdx.x;
    if (i < n) out[i] = f2bf(in[i]);
}

// ---------------------------------------------------------------------------
// Shared GEMM body: C[128x128] tile per workgroup, 8 waves x (16 rows, 8 subtiles).
// A fragments direct from global (rows are wave-exclusive); B tile staged in
// LDS via double-buffered async copies (shared by all 8 waves).
// ---------------------------------------------------------------------------
#define GEMM_BODY(A_PTR, B_PTR, MT, NT_IDX)                                            \
    __shared__ __align__(64) bf16_t bTile[2][128][32];                                 \
    const v8f vzero = {0.f,0.f,0.f,0.f,0.f,0.f,0.f,0.f};                               \
    v8f acc[8];                                                                        \
    _Pragma("unroll")                                                                  \
    for (int s = 0; s < 8; ++s) acc[s] = vzero;                                        \
    /* stage B tile [128 cols][32 k] for k-slice k0 into buffer b */                   \
    auto stage = [&](int b, int k0) {                                                  \
        _Pragma("unroll")                                                              \
        for (int c = 0; c < 2; ++c) {                                                  \
            int chunk = tid + c * 256;            /* 0..511 chunks of 16B */           \
            int col   = chunk >> 2;               /* 0..127 */                         \
            int kq    = (chunk & 3) << 3;         /* 0,8,16,24 */                      \
            const bf16_t* g = (B_PTR) + (size_t)((NT_IDX) * 128 + col) * DMODEL        \
                                       + k0 + kq;                                      \
            async_g2l_b128(g, &bTile[b][col][kq]);                                     \
        }                                                                              \
    };                                                                                 \
    stage(0, 0);                                                                       \
    size_t mrow = (size_t)((MT) * 128 + wv * 16 + l16);                                \
    for (int k0 = 0; k0 < DMODEL; k0 += 32) {                                          \
        int buf = (k0 >> 5) & 1;                                                       \
        if (k0 + 32 < DMODEL) { stage(buf ^ 1, k0 + 32); wait_async2(); }              \
        else                  { wait_async0(); }                                       \
        __syncthreads();                                                               \
        const bf16_t* ap = (A_PTR) + mrow * DMODEL + k0;                               \
        v16bf a = ld_a_pair(ap + 16 * lhalf, ap + 16 * lhalf + 8);                     \
        _Pragma("unroll")                                                              \
        for (int s = 0; s < 8; ++s) {                                                  \
            v16bf bfrag = *(const v16bf*)&bTile[buf][s * 16 + l16][16 * lhalf];        \
            acc[s] = __builtin_amdgcn_wmma_f32_16x16x32_bf16(                          \
                        false, a, false, bfrag, (short)0, acc[s], false, false);       \
        }                                                                              \
        __syncthreads();                                                               \
    }

// ---------------------------------------------------------------------------
// QKV GEMM: [65536 x 768] @ [2304 x 768]^T + bias; scatter epilogue into
// q (pre-scaled, [bn][h][tok][d]), k ([bn][h][tok][d]), v^T ([bn][h][d][tok]).
// ---------------------------------------------------------------------------
__global__ __launch_bounds__(256) void qkv_gemm_kernel(
        const bf16_t* __restrict__ xB, const bf16_t* __restrict__ wB,
        const float* __restrict__ bias,
        bf16_t* __restrict__ qO, bf16_t* __restrict__ kO, bf16_t* __restrict__ vO) {
    const int NTILES = QKV_N / 128;             // 18
    int mt = blockIdx.x / NTILES;
    int nt = blockIdx.x % NTILES;
    int tid = threadIdx.x;
    int wv = tid >> 5, lane = tid & 31, lhalf = lane >> 4, l16 = lane & 15;

    GEMM_BODY(xB, wB, mt, nt)

#pragma unroll
    for (int s = 0; s < 8; ++s) {
#pragma unroll
        for (int i = 0; i < 8; ++i) {
            int row = mt * 128 + wv * 16 + i + 8 * lhalf;   // token index
            int col = nt * 128 + s * 16 + l16;              // 0..2303
            float v = acc[s][i] + bias[col];
            int part = col / DMODEL;
            int rem  = col - part * DMODEL;
            int hh = rem / HD, dd = rem - (rem / HD) * HD;
            int bn = row >> 7, wt = row & 127;
            size_t bh = (size_t)bn * NHEAD + hh;
            if (part == 0)      qO[(bh * WTOK + wt) * HD + dd] = f2bf(v * 0.125f);
            else if (part == 1) kO[(bh * WTOK + wt) * HD + dd] = f2bf(v);
            else                vO[(bh * HD + dd) * WTOK + wt] = f2bf(v);
        }
    }
}

// ---------------------------------------------------------------------------
// Projection GEMM: [65536 x 768] @ [768 x 768]^T + bias -> f32 d_out
// ---------------------------------------------------------------------------
__global__ __launch_bounds__(256) void proj_gemm_kernel(
        const bf16_t* __restrict__ aB, const bf16_t* __restrict__ wB,
        const float* __restrict__ bias, float* __restrict__ out) {
    const int NTILES = DMODEL / 128;            // 6
    int mt = blockIdx.x / NTILES;
    int nt = blockIdx.x % NTILES;
    int tid = threadIdx.x;
    int wv = tid >> 5, lane = tid & 31, lhalf = lane >> 4, l16 = lane & 15;

    GEMM_BODY(aB, wB, mt, nt)

#pragma unroll
    for (int s = 0; s < 8; ++s) {
#pragma unroll
        for (int i = 0; i < 8; ++i) {
            int row = mt * 128 + wv * 16 + i + 8 * lhalf;
            int col = nt * 128 + s * 16 + l16;
            out[(size_t)row * DMODEL + col] = acc[s][i] + bias[col];
        }
    }
}

// ---------------------------------------------------------------------------
// Fused attention: one workgroup per (bn, h). 8 waves, each owns 16 query rows.
// K and V^T tiles staged via async copies into LDS (shared by all waves).
// S = Q@K^T (WMMA), softmax -> prune(p>0.5) -> *mask*edge_bias -> (==0 -> -1e4)
// -> softmax -> P@V (WMMA, P staged through LDS as bf16).
// ---------------------------------------------------------------------------
__global__ __launch_bounds__(256) void attn_kernel(
        const bf16_t* __restrict__ qB, const bf16_t* __restrict__ kB,
        const bf16_t* __restrict__ vB, const float* __restrict__ mask,
        const float* __restrict__ ebias, bf16_t* __restrict__ outB) {
    int bh = blockIdx.x;
    int bn = bh / NHEAD, h = bh % NHEAD;
    int fw = bn & (FW - 1);
    int tid = threadIdx.x;
    int wv = tid >> 5, lane = tid & 31, lhalf = lane >> 4, l16 = lane & 15;

    __shared__ __align__(64) bf16_t kLds[WTOK][HD];     // 16 KB, [tok][d]
    __shared__ __align__(64) bf16_t vLds[HD][WTOK];     // 16 KB, [d][tok]
    __shared__ __align__(64) bf16_t pLds[WTOK][WTOK];   // 32 KB probabilities

    const bf16_t* qbase = qB + (size_t)bh * WTOK * HD;
    const bf16_t* kbase = kB + (size_t)bh * WTOK * HD;
    const bf16_t* vbase = vB + (size_t)bh * HD * WTOK;

    // --- async-stage K and V^T (flat layouts match LDS layouts exactly)
    bf16_t* kFlat = &kLds[0][0];
    bf16_t* vFlat = &vLds[0][0];
#pragma unroll
    for (int c = 0; c < 8; ++c) {
        int chunk = tid + c * 256;                 // 0..2047 chunks of 8 bf16
        if (chunk < 1024) {
            int off = chunk * 8;
            async_g2l_b128(kbase + off, kFlat + off);
        } else {
            int off = (chunk - 1024) * 8;
            async_g2l_b128(vbase + off, vFlat + off);
        }
    }

    const v8f vzero = {0.f,0.f,0.f,0.f,0.f,0.f,0.f,0.f};
    v8f acc[8];
#pragma unroll
    for (int t = 0; t < 8; ++t) acc[t] = vzero;

    wait_async0();
    __syncthreads();

    // --- S = Q @ K^T for rows [16*wv, 16*wv+16), K from LDS
    int ma = 16 * wv + l16;
#pragma unroll
    for (int k0 = 0; k0 < HD; k0 += 32) {
        const bf16_t* ap = qbase + ma * HD + k0;
        v16bf a = ld_a_pair(ap + 16 * lhalf, ap + 16 * lhalf + 8);
#pragma unroll
        for (int t = 0; t < 8; ++t) {
            v16bf b = *(const v16bf*)&kLds[16 * t + l16][k0 + 16 * lhalf];
            acc[t] = __builtin_amdgcn_wmma_f32_16x16x32_bf16(
                        false, a, false, b, (short)0, acc[t], false, false);
        }
    }

    // --- two-pass softmax with pruning. C-layout: elem i of lane l is
    // row (i + 8*(l>>4)), col (16*t + (l&15)); a full row lives across one
    // 16-lane half, so reduce with xor-shuffles 1,2,4,8.
#pragma unroll
    for (int i = 0; i < 8; ++i) {
        int q = 16 * wv + i + 8 * lhalf;
        float mx = -3.0e38f;
#pragma unroll
        for (int t = 0; t < 8; ++t) mx = fmaxf(mx, acc[t][i]);
#pragma unroll
        for (int d = 1; d < 16; d <<= 1) mx = fmaxf(mx, __shfl_xor(mx, d, 32));
        float s = 0.f;
#pragma unroll
        for (int t = 0; t < 8; ++t) s += __expf(acc[t][i] - mx);
#pragma unroll
        for (int d = 1; d < 16; d <<= 1) s += __shfl_xor(s, d, 32);
        float inv = 1.f / s;

        const float* mrow = mask  + ((size_t)fw * WTOK + q) * WTOK;
        const float* brow = ebias + ((size_t)h  * WTOK + q) * WTOK;

        float vals[8];
        float nmx = -3.0e38f;
#pragma unroll
        for (int t = 0; t < 8; ++t) {
            int kc = 16 * t + l16;
            float p = __expf(acc[t][i] - mx) * inv;
            float a = (p > 0.5f) ? 0.f : acc[t][i];       // prune
            a = a * mrow[kc] * brow[kc];                   // mask * edge_bias
            a = (a == 0.f) ? -10000.f : a;                 // kill zeros
            vals[t] = a;
            nmx = fmaxf(nmx, a);
        }
#pragma unroll
        for (int d = 1; d < 16; d <<= 1) nmx = fmaxf(nmx, __shfl_xor(nmx, d, 32));
        float s2 = 0.f;
#pragma unroll
        for (int t = 0; t < 8; ++t) { vals[t] = __expf(vals[t] - nmx); s2 += vals[t]; }
#pragma unroll
        for (int d = 1; d < 16; d <<= 1) s2 += __shfl_xor(s2, d, 32);
        float inv2 = 1.f / s2;
#pragma unroll
        for (int t = 0; t < 8; ++t)
            pLds[q][16 * t + l16] = f2bf(vals[t] * inv2);
    }
    __syncthreads();

    // --- O = P @ V  (A from pLds, B = V^T from vLds)
    v8f oacc[4];
#pragma unroll
    for (int t = 0; t < 4; ++t) oacc[t] = vzero;
#pragma unroll
    for (int kk = 0; kk < WTOK; kk += 32) {
        const bf16_t* ap = &pLds[16 * wv + l16][kk];
        v16bf a = ld_a_pair(ap + 16 * lhalf, ap + 16 * lhalf + 8);
#pragma unroll
        for (int t = 0; t < 4; ++t) {
            v16bf b = *(const v16bf*)&vLds[16 * t + l16][kk + 16 * lhalf];
            oacc[t] = __builtin_amdgcn_wmma_f32_16x16x32_bf16(
                        false, a, false, b, (short)0, oacc[t], false, false);
        }
    }

    // store O in [Bn, W, D] bf16 layout (D index = h*64 + d)
#pragma unroll
    for (int t = 0; t < 4; ++t) {
#pragma unroll
        for (int i = 0; i < 8; ++i) {
            int q = 16 * wv + i + 8 * lhalf;
            int d = 16 * t + l16;
            outB[((size_t)bn * WTOK + q) * DMODEL + h * HD + d] = f2bf(oacc[t][i]);
        }
    }
}

// ---------------------------------------------------------------------------
extern "C" void kernel_launch(void* const* d_in, const int* in_sizes, int n_in,
                              void* d_out, int out_size, void* d_ws, size_t ws_size,
                              hipStream_t stream) {
    const float* x      = (const float*)d_in[0];   // [512,128,768]
    const float* qkv_w  = (const float*)d_in[1];   // [2304,768]
    const float* qkv_b  = (const float*)d_in[2];   // [2304]
    const float* proj_w = (const float*)d_in[3];   // [768,768]
    const float* proj_b = (const float*)d_in[4];   // [768]
    const float* ebias  = (const float*)d_in[5];   // [1,12,128,128]
    const float* mask   = (const float*)d_in[6];   // [128,128,128]

    const size_t M      = (size_t)BN * WTOK;        // 65536
    const size_t xN     = M * DMODEL;               // 50,331,648
    const size_t wqN    = (size_t)QKV_N * DMODEL;   // 1,769,472
    const size_t wpN    = (size_t)DMODEL * DMODEL;  // 589,824
    const size_t headN  = (size_t)BN * NHEAD * WTOK * HD;  // per q/k/v buffer

    char* ws = (char*)d_ws;
    bf16_t* xB  = (bf16_t*)ws;                 ws += xN   * sizeof(bf16_t);
    bf16_t* wqB = (bf16_t*)ws;                 ws += wqN  * sizeof(bf16_t);
    bf16_t* wpB = (bf16_t*)ws;                 ws += wpN  * sizeof(bf16_t);
    bf16_t* qO  = (bf16_t*)ws;                 ws += headN * sizeof(bf16_t);
    bf16_t* kO  = (bf16_t*)ws;                 ws += headN * sizeof(bf16_t);
    bf16_t* vO  = (bf16_t*)ws;                 ws += headN * sizeof(bf16_t);
    bf16_t* oB  = (bf16_t*)ws;                 ws += xN   * sizeof(bf16_t);

    cvt_f32_bf16<<<(unsigned)((xN  + 255) / 256), 256, 0, stream>>>(x,      xB,  (int)xN);
    cvt_f32_bf16<<<(unsigned)((wqN + 255) / 256), 256, 0, stream>>>(qkv_w,  wqB, (int)wqN);
    cvt_f32_bf16<<<(unsigned)((wpN + 255) / 256), 256, 0, stream>>>(proj_w, wpB, (int)wpN);

    qkv_gemm_kernel<<<(M / 128) * (QKV_N / 128), 256, 0, stream>>>(
        xB, wqB, qkv_b, qO, kO, vO);

    attn_kernel<<<BN * NHEAD, 256, 0, stream>>>(
        qO, kO, vO, mask, ebias, oB);

    proj_gemm_kernel<<<(M / 128) * (DMODEL / 128), 256, 0, stream>>>(
        oB, wpB, proj_b, (float*)d_out);
}
